// GraphModelGGNN_3272765080010
// MI455X (gfx1250) — compile-verified, matrix operand order checked by reading, not run on hardware
//
#include <hip/hip_runtime.h>

// ---------------------------------------------------------------------------
// GGNN on MI455X (gfx1250, wave32, WMMA).
//
//  * All GEMMs: bf16 WMMA (v_wmma_f32_16x16x32_bf16), f32 accumulate, K=128.
//  * Each wave computes a 16x128 row strip (8 C tiles): A fragment loaded
//    once per K-step, amortized over 8 WMMAs -> 8x less A traffic.
//  * Per-edge-type transform restructured as R dense GEMMs over all nodes
//    (halves FLOPs vs per-edge transform); edge work is then a pure
//    bandwidth stream: wave/edge, 4 global_atomic_add_f32 per lane.
//  * Epilogue variants are compile-time template parameters (no runtime
//    branches around WMMA/stores).
// ---------------------------------------------------------------------------

typedef __bf16 bf16_t;
typedef __attribute__((ext_vector_type(16))) __bf16 v16bf;
typedef __attribute__((ext_vector_type(8)))  __bf16 v8bf;
typedef __attribute__((ext_vector_type(8)))  float  v8f;

union Frag { v16bf v; v8bf h[2]; };

__device__ __forceinline__ bf16_t f2bf(float f) {
  union { float f; unsigned int u; } v; v.f = f;
  unsigned int r = (v.u + 0x7fffu + ((v.u >> 16) & 1u)) >> 16;  // RNE
  union { unsigned short s; bf16_t b; } o; o.s = (unsigned short)r;
  return o.b;
}

// Problem constants
#define BB 8
#define NN 4096
#define EE 65536
#define HH 128
#define RR 8
#define KITER 3
#define MM (BB * NN)      // 32768 flattened nodes

// ---------------------------------------------------------------------------
// bf16 WMMA GEMM:  out[M,N] = act(A[M,128] @ W[N,128]^T + bias) * rowmask
// Block = 128 threads (4 waves). Wave w computes rows m0..m0+15 with
// m0 = (blockIdx.y*4 + w)*16, across a 128-column strip nstrip =
// blockIdx.x*128 (8 accumulator tiles). Grids are exact multiples of the
// tile sizes -> EXEC is all ones at every WMMA (ISA requirement).
//
// Fragment layouts (ISA 7.12.2, 16-bit 16x32 A):
//   lanes 0-15:  row M = lane,    K = k0+[0..7] (v0-3), k0+16+[0..7] (v4-7)
//   lanes 16-31: row M = lane-16, K shifted by +8.
// B[k][n] = W[n][k] (row-major W) -> symmetric build with N in place of M,
// so both fragments are two aligned 16-byte global loads per lane.
// ---------------------------------------------------------------------------
template <int RELU, int MASK, int OUTF, int OUTB>
__global__ void gemm_bf16_kernel(const bf16_t* __restrict__ A,
                                 const bf16_t* __restrict__ W,
                                 const float* __restrict__ bias,
                                 float* __restrict__ outF,
                                 bf16_t* __restrict__ outB,
                                 const float* __restrict__ rowmask,
                                 int N) {
  const int lane   = threadIdx.x & 31;
  const int wave   = threadIdx.x >> 5;
  const int m0     = (blockIdx.y * 4 + wave) * 16;
  const int nstrip = blockIdx.x * 128;
  const int half   = lane >> 4;
  const int l15    = lane & 15;

  // Accumulators seeded with per-column bias (C layout: col = tile_n0 + lane%16).
  v8f c[8];
#pragma unroll
  for (int nt = 0; nt < 8; ++nt) {
    const float bv = bias[nstrip + nt * 16 + l15];
#pragma unroll
    for (int i = 0; i < 8; ++i) c[nt][i] = bv;
  }

  const bf16_t* pa0 = A + (m0 + l15) * 128 + half * 8;
  const bf16_t* pb0 = W + (nstrip + l15) * 128 + half * 8;

#pragma unroll
  for (int k0 = 0; k0 < 128; k0 += 32) {
    Frag a;
    a.h[0] = *(const v8bf*)(pa0 + k0);
    a.h[1] = *(const v8bf*)(pa0 + k0 + 16);
#pragma unroll
    for (int nt = 0; nt < 8; ++nt) {
      Frag b;
      const bf16_t* pb = pb0 + nt * (16 * 128) + k0;
      b.h[0] = *(const v8bf*)pb;
      b.h[1] = *(const v8bf*)(pb + 16);
      c[nt] = __builtin_amdgcn_wmma_f32_16x16x32_bf16(
                  /*neg_a=*/false, a.v, /*neg_b=*/false, b.v,
                  /*c_mod=*/(short)0, c[nt], /*reuse_a=*/false, /*reuse_b=*/false);
    }
  }

  // Epilogue. C layout: VGPR i -> row m0+i (lanes 0-15) / m0+8+i (lanes 16-31).
  float rm[8];
#pragma unroll
  for (int i = 0; i < 8; ++i)
    rm[i] = MASK ? rowmask[m0 + i + half * 8] : 1.0f;

#pragma unroll
  for (int nt = 0; nt < 8; ++nt) {
    const int col = nstrip + nt * 16 + l15;
#pragma unroll
    for (int i = 0; i < 8; ++i) {
      const int row = m0 + i + half * 8;
      float v = c[nt][i];
      if (RELU) v = v > 0.0f ? v : 0.0f;
      if (MASK) v *= rm[i];
      if (OUTF) outF[row * N + col] = v;
      if (OUTB) outB[row * N + col] = f2bf(v);
    }
  }
}

// ---------------------------------------------------------------------------
// Node encoder: x0 = relu(concat(class_emb[class], states @ state_W^T + b))
// One block (128 threads) per node; writes bf16 directly for the first GEMM.
// ---------------------------------------------------------------------------
__global__ void enc0_kernel(const int* __restrict__ cls,
                            const float* __restrict__ states,
                            const float* __restrict__ class_emb,
                            const float* __restrict__ state_W,
                            const float* __restrict__ state_b,
                            bf16_t* __restrict__ x0b) {
  const int m = blockIdx.x;
  const int j = threadIdx.x;   // 0..127
  float v;
  if (j < 64) {
    v = class_emb[cls[m] * 64 + j];
  } else {
    const int jj = j - 64;
    float acc = state_b[jj];
    const float* st = states + m * 16;
    const float* w  = state_W + jj * 16;
#pragma unroll
    for (int s = 0; s < 16; ++s) acc += w[s] * st[s];
    v = acc;
  }
  v = v > 0.0f ? v : 0.0f;
  x0b[m * HH + j] = f2bf(v);
}

// ---------------------------------------------------------------------------
// Edge scatter: one wave per edge; edges of type rsel add tmp[src]*em into
// a[dst] (128 floats = 4 dwords/lane) via native f32 global atomics.
// ---------------------------------------------------------------------------
__global__ void scatter_kernel(const float* __restrict__ tmp,
                               const int* __restrict__ etup,
                               const int* __restrict__ ecls,
                               const float* __restrict__ emask,
                               float* __restrict__ a, int rsel) {
  const int gid  = blockIdx.x * blockDim.x + threadIdx.x;
  const int e    = gid >> 5;          // global edge index in [0, B*E)
  const int lane = gid & 31;
  if (ecls[e] != rsel) return;
  const int b   = e >> 16;            // e / EE (EE = 65536)
  const int src = etup[e * 2 + 0] + b * NN;
  const int dst = etup[e * 2 + 1] + b * NN;
  const float em = emask[e];
  const float* ts = tmp + src * HH + lane;
  float* ad = a + dst * HH + lane;
  __builtin_prefetch(ts, 0, 0);       // global_prefetch_b8: pull message row
#pragma unroll
  for (int j = 0; j < 4; ++j)
    unsafeAtomicAdd(ad + j * 32, ts[j * 32] * em);
}

// ---------------------------------------------------------------------------
// GRU gate combine (torch GRUCell order r,z,n):
//   r = sig(gi[:,0:128]+gh[:,0:128]); z = sig(gi[:,128:256]+gh[:,128:256])
//   n = tanh(gi[:,256:384] + r*gh[:,256:384]); h' = ((1-z)n + z h) * nmask
// ---------------------------------------------------------------------------
__global__ void gru_kernel(const float* __restrict__ gi,
                           const float* __restrict__ gh,
                           const float* __restrict__ h_in,
                           const float* __restrict__ nmask,
                           float* __restrict__ h_out,
                           bf16_t* __restrict__ hb_out,
                           float* __restrict__ out2) {
  const int idx = blockIdx.x * blockDim.x + threadIdx.x;  // M*128
  const int m = idx >> 7;
  const int j = idx & 127;
  const float* gim = gi + m * 384;
  const float* ghm = gh + m * 384;
  const float ir = gim[j], iz = gim[j + 128], in = gim[j + 256];
  const float hr = ghm[j], hz = ghm[j + 128], hn = ghm[j + 256];
  const float r = 1.0f / (1.0f + __expf(-(ir + hr)));
  const float z = 1.0f / (1.0f + __expf(-(iz + hz)));
  const float n = tanhf(in + r * hn);
  const float h = h_in[idx];
  const float hnew = ((1.0f - z) * n + z * h) * nmask[m];
  h_out[idx] = hnew;
  hb_out[idx] = f2bf(hnew);
  if (out2) out2[idx] = hnew;
}

// ---------------------------------------------------------------------------
// Utility kernels
// ---------------------------------------------------------------------------
__global__ void cvt_f32_bf16_kernel(const float* __restrict__ src,
                                    bf16_t* __restrict__ dst, int n) {
  const int i = blockIdx.x * blockDim.x + threadIdx.x;
  if (i < n) dst[i] = f2bf(src[i]);
}

__global__ void zero_f32_kernel(float* __restrict__ p, int n) {
  const int i = blockIdx.x * blockDim.x + threadIdx.x;
  if (i < n) p[i] = 0.0f;
}

// ---------------------------------------------------------------------------
extern "C" void kernel_launch(void* const* d_in, const int* in_sizes, int n_in,
                              void* d_out, int out_size, void* d_ws, size_t ws_size,
                              hipStream_t stream) {
  const int*   class_objects = (const int*)  d_in[0];
  const float* states        = (const float*)d_in[1];
  const int*   edge_tuples   = (const int*)  d_in[2];
  const int*   edge_classes  = (const int*)  d_in[3];
  const float* mask_object   = (const float*)d_in[4];
  const float* mask_edge     = (const float*)d_in[5];
  const float* class_emb     = (const float*)d_in[6];
  const float* state_W       = (const float*)d_in[7];
  const float* state_b       = (const float*)d_in[8];
  const float* W1            = (const float*)d_in[9];
  const float* b1            = (const float*)d_in[10];
  const float* W2            = (const float*)d_in[11];
  const float* b2            = (const float*)d_in[12];
  const float* Ws            = (const float*)d_in[13];
  const float* bs            = (const float*)d_in[14];
  const float* gru_Wih       = (const float*)d_in[15];
  const float* gru_Whh       = (const float*)d_in[16];
  const float* gru_bih       = (const float*)d_in[17];
  const float* gru_bhh       = (const float*)d_in[18];
  float* out = (float*)d_out;

  // Workspace carve-up
  char* base = (char*)d_ws;
  size_t off = 0;
  auto carve = [&](size_t bytes) {
    void* p = base + off;
    off += (bytes + 255) & ~(size_t)255;
    return p;
  };
  float*  bh   = (float*) carve(sizeof(float)  * MM * HH);   // current h (f32)
  float*  ba   = (float*) carve(sizeof(float)  * MM * HH);   // aggregated msgs
  float*  btmp = (float*) carve(sizeof(float)  * MM * HH);   // per-type transform
  float*  bgi  = (float*) carve(sizeof(float)  * MM * 384);  // GRU input gates
  float*  bgh  = (float*) carve(sizeof(float)  * MM * 384);  // GRU hidden gates
  bf16_t* xb0  = (bf16_t*)carve(sizeof(bf16_t) * MM * HH);   // bf16 ping
  bf16_t* xb1  = (bf16_t*)carve(sizeof(bf16_t) * MM * HH);   // bf16 pong
  bf16_t* W1b  = (bf16_t*)carve(sizeof(bf16_t) * HH * HH);
  bf16_t* W2b  = (bf16_t*)carve(sizeof(bf16_t) * HH * HH);
  bf16_t* Wsb  = (bf16_t*)carve(sizeof(bf16_t) * RR * HH * HH);
  bf16_t* Wihb = (bf16_t*)carve(sizeof(bf16_t) * 3 * HH * HH);
  bf16_t* Whhb = (bf16_t*)carve(sizeof(bf16_t) * 3 * HH * HH);
  (void)ws_size; (void)in_sizes; (void)n_in; (void)out_size;

  // Weight conversion to bf16 (tiny; deterministic each call)
  cvt_f32_bf16_kernel<<<(HH*HH + 255)/256,   256, 0, stream>>>(W1, W1b, HH*HH);
  cvt_f32_bf16_kernel<<<(HH*HH + 255)/256,   256, 0, stream>>>(W2, W2b, HH*HH);
  cvt_f32_bf16_kernel<<<(RR*HH*HH + 255)/256,256, 0, stream>>>(Ws, Wsb, RR*HH*HH);
  cvt_f32_bf16_kernel<<<(3*HH*HH + 255)/256, 256, 0, stream>>>(gru_Wih, Wihb, 3*HH*HH);
  cvt_f32_bf16_kernel<<<(3*HH*HH + 255)/256, 256, 0, stream>>>(gru_Whh, Whhb, 3*HH*HH);

  const dim3 blk(128);
  const dim3 grid128(1, MM / 64);   // N=128: 1 column strip, 4 M-tiles/block
  const dim3 grid384(3, MM / 64);   // N=384: 3 column strips

  // Node encoder: x0 -> x1 -> h (mask applied to f32 and bf16 outputs)
  enc0_kernel<<<MM, 128, 0, stream>>>(class_objects, states, class_emb,
                                      state_W, state_b, xb0);
  gemm_bf16_kernel<1,0,0,1><<<grid128, blk, 0, stream>>>(
      xb0, W1b, b1, nullptr, xb1, nullptr, 128);
  gemm_bf16_kernel<1,1,1,1><<<grid128, blk, 0, stream>>>(
      xb1, W2b, b2, bh, xb0, mask_object, 128);
  // xb0 now holds bf16 of masked h.

  const int nElems = MM * HH;
  for (int k = 0; k < KITER; ++k) {
    zero_f32_kernel<<<nElems / 256, 256, 0, stream>>>(ba, nElems);
    for (int r = 0; r < RR; ++r) {
      // tmp = h @ Ws[r]^T + bs[r]  (dense WMMA GEMM over all nodes)
      gemm_bf16_kernel<0,0,1,0><<<grid128, blk, 0, stream>>>(
          xb0, Wsb + r * HH * HH, bs + r * HH, btmp, nullptr, nullptr, 128);
      // edges of type r: a[dst] += tmp[src] * edge_mask
      scatter_kernel<<<(BB * EE * 32) / 256, 256, 0, stream>>>(
          btmp, edge_tuples, edge_classes, mask_edge, ba, r);
    }
    // GRU: gi = a @ Wih^T + bih ; gh = h @ Whh^T + bhh
    cvt_f32_bf16_kernel<<<nElems / 256, 256, 0, stream>>>(ba, xb1, nElems);
    gemm_bf16_kernel<0,0,1,0><<<grid384, blk, 0, stream>>>(
        xb1, Wihb, gru_bih, bgi, nullptr, nullptr, 384);
    gemm_bf16_kernel<0,0,1,0><<<grid384, blk, 0, stream>>>(
        xb0, Whhb, gru_bhh, bgh, nullptr, nullptr, 384);
    gru_kernel<<<nElems / 256, 256, 0, stream>>>(
        bgi, bgh, bh, mask_object, bh, xb0,
        (k == KITER - 1) ? out : nullptr);
  }
}
